// MultiHeadAttention_4269197492560
// MI455X (gfx1250) — matrix operand order repaired and checked by reading, not compiled, not run on hardware
//
#include <hip/hip_runtime.h>

// ---------------------------------------------------------------------------
// MultiHeadAttention for MI455X (gfx1250), bf16 WMMA pipeline.
// B=2, N=2048, DIM=1024, HEADS=16, DH=64
// Stages:
//   1) cast x, w_qkv, w_out  fp32 -> bf16
//   2) qkv GEMM (bf16 WMMA, 32x64 tile/wave), scatter into q/k/vT head layouts
//   3) flash attention: one wave per (b,h,32-row tile), streaming softmax
//   4) output GEMM (bf16 WMMA, 32x64 tile/wave) + bias -> fp32 d_out
// ---------------------------------------------------------------------------

typedef __bf16 bf16_t;
typedef __attribute__((ext_vector_type(16))) __bf16 v16bf;
typedef __attribute__((ext_vector_type(8)))  __bf16 v8bf;
typedef __attribute__((ext_vector_type(8)))  float  v8f;

#define BATCH 2
#define NSEQ  2048
#define DMODEL 1024
#define NHEADS 16
#define DHEAD  64
#define MTOT   (BATCH * NSEQ)   // 4096
#define SCALE  0.125f           // 64^-0.5

static __device__ __forceinline__ bf16_t f2bf(float f) {
    unsigned int u = __float_as_uint(f);
    unsigned int r = (u + 0x7FFFu + ((u >> 16) & 1u)) >> 16;
    unsigned short s = (unsigned short)r;
    return __builtin_bit_cast(bf16_t, s);
}

static __device__ __forceinline__ v16bf pack16(v8bf lo, v8bf hi) {
    v16bf a;
#pragma unroll
    for (int i = 0; i < 8; ++i) { a[i] = lo[i]; a[8 + i] = hi[i]; }
    return a;
}

static __device__ __forceinline__ v8f wmma_bf16(v16bf a, v16bf b, v8f c) {
    return __builtin_amdgcn_wmma_f32_16x16x32_bf16(
        /*neg_a=*/false, a, /*neg_b=*/false, b,
        /*c_mod=*/(short)0, c, /*reuse_a=*/false, /*reuse_b=*/false);
}

// ---------------------------------------------------------------------------
// Stage 1: fp32 -> bf16 cast
// ---------------------------------------------------------------------------
__global__ void cast_f32_bf16(const float* __restrict__ src,
                              bf16_t* __restrict__ dst, int n) {
    int i = blockIdx.x * blockDim.x + threadIdx.x;
    int stride = gridDim.x * blockDim.x;
    for (; i < n; i += stride) dst[i] = f2bf(src[i]);
}

// ---------------------------------------------------------------------------
// Stage 2: qkv = x @ w_qkv^T  (M=4096, N=3072, K=1024)
// One wave per 32x64 output tile: two A-fragments share four B-fragments
// per k-step -> 8 WMMAs / 12 b128 loads. Scatters into head layouts; V is
// stored transposed per head so the PV GEMM gets contiguous B-fragments.
// ---------------------------------------------------------------------------
__global__ __launch_bounds__(32)
void qkv_gemm(const bf16_t* __restrict__ X,     // [4096,1024]
              const bf16_t* __restrict__ W,     // [3072,1024]
              bf16_t* __restrict__ Qo,          // [b,h,n,d]
              bf16_t* __restrict__ Ko,          // [b,h,n,d]
              bf16_t* __restrict__ Vt) {        // [b,h,d,n]
    const int lane = threadIdx.x & 31;
    const int half = lane >> 4;
    const int l15  = lane & 15;
    const int m0 = blockIdx.x * 32;
    const int n0 = blockIdx.y * 64;

    v8f acc[2][4] = {};
    const bf16_t* arow0 = X + (size_t)(m0 + l15) * DMODEL + half * 16;
    const bf16_t* arow1 = X + (size_t)(m0 + 16 + l15) * DMODEL + half * 16;

    for (int k0 = 0; k0 < DMODEL; k0 += 32) {
        if (k0 + 32 < DMODEL) {
            __builtin_prefetch(arow0 + k0 + 32, 0, 3);
            __builtin_prefetch(arow1 + k0 + 32, 0, 3);
        }
        v16bf a0 = pack16(*(const v8bf*)(arow0 + k0),
                          *(const v8bf*)(arow0 + k0 + 8));
        v16bf a1 = pack16(*(const v8bf*)(arow1 + k0),
                          *(const v8bf*)(arow1 + k0 + 8));
#pragma unroll
        for (int t = 0; t < 4; ++t) {
            const bf16_t* brow =
                W + (size_t)(n0 + t * 16 + l15) * DMODEL + k0 + half * 16;
            v16bf b = *(const v16bf*)brow;
            acc[0][t] = wmma_bf16(a0, b, acc[0][t]);
            acc[1][t] = wmma_bf16(a1, b, acc[1][t]);
        }
    }

#pragma unroll
    for (int rt = 0; rt < 2; ++rt) {
#pragma unroll
        for (int t = 0; t < 4; ++t) {
#pragma unroll
            for (int r = 0; r < 8; ++r) {
                int row = m0 + rt * 16 + r + half * 8;  // token row (0..4095)
                int e   = n0 + t * 16 + l15;            // column in [0,3072)
                int bb  = row >> 11;                    // /2048
                int tok = row & (NSEQ - 1);
                int sec = e >> 10;                      // 0=q 1=k 2=v
                int eh  = e & (DMODEL - 1);
                int h   = eh >> 6;
                int d   = eh & (DHEAD - 1);
                bf16_t v = f2bf(acc[rt][t][r]);
                size_t head = (size_t)(bb * NHEADS + h);
                if (sec == 0)
                    Qo[(head * NSEQ + tok) * DHEAD + d] = v;
                else if (sec == 1)
                    Ko[(head * NSEQ + tok) * DHEAD + d] = v;
                else
                    Vt[(head * DHEAD + d) * NSEQ + tok] = v;
            }
        }
    }
}

// ---------------------------------------------------------------------------
// Stage 3: flash attention. One wave per (b, h, 32-row tile).
// Two 16-row sub-tiles share every K/V fragment. O (2 x 16x64 fp32, 64 VGPRs)
// stays in registers; running max/sum per row via shfl_xor reductions across
// the 16-lane groups of the WMMA C layout. P round-trips 2KB of LDS to
// convert C-layout -> A-layout.
// ---------------------------------------------------------------------------
__global__ __launch_bounds__(32)
void attn_kernel(const bf16_t* __restrict__ Q,   // [b,h,n,d]
                 const bf16_t* __restrict__ K,   // [b,h,n,d]
                 const bf16_t* __restrict__ Vt,  // [b,h,d,n]
                 bf16_t* __restrict__ AO) {      // [b*n, 1024]
    __shared__ bf16_t plds[32 * 32];   // P tiles staging (two 16x32 tiles)

    const int lane = threadIdx.x & 31;
    const int half = lane >> 4;
    const int l15  = lane & 15;
    const int it = blockIdx.x * 32;
    const int h  = blockIdx.y;
    const int b  = blockIdx.z;

    const size_t headoff = (size_t)(b * NHEADS + h) * NSEQ * DHEAD;
    const bf16_t* Qh = Q + headoff;
    const bf16_t* Kh = K + headoff;
    const bf16_t* Vh = Vt + headoff;   // [64][2048]

    // Q A-fragments: [sub-tile][d-half], rows it..it+15 and it+16..it+31
    v16bf qf[2][2];
#pragma unroll
    for (int rt = 0; rt < 2; ++rt) {
        const bf16_t* qrow = Qh + (size_t)(it + rt * 16 + l15) * DHEAD + half * 16;
        qf[rt][0] = pack16(*(const v8bf*)(qrow),      *(const v8bf*)(qrow + 8));
        qf[rt][1] = pack16(*(const v8bf*)(qrow + 32), *(const v8bf*)(qrow + 40));
    }

    float mrun[2][8], lrun[2][8];
#pragma unroll
    for (int rt = 0; rt < 2; ++rt)
#pragma unroll
        for (int r = 0; r < 8; ++r) { mrun[rt][r] = -1e30f; lrun[rt][r] = 0.0f; }
    v8f o[2][4] = {};

    for (int j0 = 0; j0 < NSEQ; j0 += 32) {
        // ---- K B-fragments for 32 columns (shared by both sub-tiles) ----
        const bf16_t* kr0 = Kh + (size_t)(j0 + l15) * DHEAD + half * 16;
        const bf16_t* kr1 = Kh + (size_t)(j0 + 16 + l15) * DHEAD + half * 16;
        if (j0 + 32 < NSEQ) {
            __builtin_prefetch(kr0 + 32 * DHEAD, 0, 3);
            __builtin_prefetch(kr1 + 32 * DHEAD, 0, 3);
        }
        v16bf kb0 = *(const v16bf*)(kr0);        // cols j0..+15,  d = 0..31
        v16bf kb1 = *(const v16bf*)(kr0 + 32);   //                d = 32..63
        v16bf kb2 = *(const v16bf*)(kr1);        // cols j0+16..+31
        v16bf kb3 = *(const v16bf*)(kr1 + 32);

#pragma unroll
        for (int rt = 0; rt < 2; ++rt) {
            // ---- S = Q K^T : two 16x16 C tiles ----
            v8f s0 = {}, s1 = {};
            s0 = wmma_bf16(qf[rt][0], kb0, s0);
            s0 = wmma_bf16(qf[rt][1], kb1, s0);
            s1 = wmma_bf16(qf[rt][0], kb2, s1);
            s1 = wmma_bf16(qf[rt][1], kb3, s1);

            // ---- streaming softmax stats (row r lives across 16 lanes) ----
            float alpha[8];
#pragma unroll
            for (int r = 0; r < 8; ++r) {
                float a0 = s0[r] * SCALE;
                float a1 = s1[r] * SCALE;
                float vmax = fmaxf(a0, a1);
#pragma unroll
                for (int mask = 1; mask < 16; mask <<= 1)
                    vmax = fmaxf(vmax, __shfl_xor(vmax, mask, 32));
                float mnew = fmaxf(mrun[rt][r], vmax);
                float al   = __expf(mrun[rt][r] - mnew);
                float p0   = __expf(a0 - mnew);
                float p1   = __expf(a1 - mnew);
                float rs   = p0 + p1;
#pragma unroll
                for (int mask = 1; mask < 16; mask <<= 1)
                    rs += __shfl_xor(rs, mask, 32);
                lrun[rt][r] = lrun[rt][r] * al + rs;
                mrun[rt][r] = mnew;
                alpha[r] = al;
                s0[r] = p0;
                s1[r] = p1;
            }

            // ---- rescale O accumulators ----
#pragma unroll
            for (int t = 0; t < 4; ++t)
#pragma unroll
                for (int r = 0; r < 8; ++r) o[rt][t][r] *= alpha[r];

            // ---- P (C layout) -> LDS ----
#pragma unroll
            for (int r = 0; r < 8; ++r) {
                int row = rt * 16 + r + half * 8;
                plds[row * 32 + l15]      = f2bf(s0[r]);
                plds[row * 32 + 16 + l15] = f2bf(s1[r]);
            }
        }
        __syncthreads();

        // ---- O += P @ V (V transposed: contiguous B-fragment loads) ----
#pragma unroll
        for (int rt = 0; rt < 2; ++rt) {
            const bf16_t* pr = &plds[(rt * 16 + l15) * 32 + half * 16];
            v16bf pf = pack16(*(const v8bf*)(pr), *(const v8bf*)(pr + 8));
#pragma unroll
            for (int t = 0; t < 4; ++t) {
                const bf16_t* vr =
                    Vh + (size_t)(t * 16 + l15) * NSEQ + j0 + half * 16;
                v16bf vf = *(const v16bf*)vr;
                o[rt][t] = wmma_bf16(pf, vf, o[rt][t]);
            }
        }
        __syncthreads();
    }

    // ---- normalize and merge heads into [b*n, 1024] bf16 ----
#pragma unroll
    for (int rt = 0; rt < 2; ++rt) {
#pragma unroll
        for (int r = 0; r < 8; ++r) {
            float inv = 1.0f / lrun[rt][r];
            int tok = it + rt * 16 + r + half * 8;
            size_t orow = ((size_t)(b * NSEQ + tok)) * DMODEL + h * DHEAD;
#pragma unroll
            for (int t = 0; t < 4; ++t)
                AO[orow + t * 16 + l15] = f2bf(o[rt][t][r] * inv);
        }
    }
}

// ---------------------------------------------------------------------------
// Stage 4: out = AO @ w_out^T + b_out  (M=4096, N=1024, K=1024), fp32 output
// Same 32x64-per-wave tiling as stage 2.
// ---------------------------------------------------------------------------
__global__ __launch_bounds__(32)
void out_gemm(const bf16_t* __restrict__ AO,     // [4096,1024]
              const bf16_t* __restrict__ W,      // [1024,1024]
              const float* __restrict__ bias,    // [1024]
              float* __restrict__ out) {         // [4096,1024] fp32
    const int lane = threadIdx.x & 31;
    const int half = lane >> 4;
    const int l15  = lane & 15;
    const int m0 = blockIdx.x * 32;
    const int n0 = blockIdx.y * 64;

    v8f acc[2][4] = {};
    const bf16_t* arow0 = AO + (size_t)(m0 + l15) * DMODEL + half * 16;
    const bf16_t* arow1 = AO + (size_t)(m0 + 16 + l15) * DMODEL + half * 16;

    for (int k0 = 0; k0 < DMODEL; k0 += 32) {
        if (k0 + 32 < DMODEL) {
            __builtin_prefetch(arow0 + k0 + 32, 0, 3);
            __builtin_prefetch(arow1 + k0 + 32, 0, 3);
        }
        v16bf a0 = pack16(*(const v8bf*)(arow0 + k0),
                          *(const v8bf*)(arow0 + k0 + 8));
        v16bf a1 = pack16(*(const v8bf*)(arow1 + k0),
                          *(const v8bf*)(arow1 + k0 + 8));
#pragma unroll
        for (int t = 0; t < 4; ++t) {
            const bf16_t* brow =
                W + (size_t)(n0 + t * 16 + l15) * DMODEL + k0 + half * 16;
            v16bf b = *(const v16bf*)brow;
            acc[0][t] = wmma_bf16(a0, b, acc[0][t]);
            acc[1][t] = wmma_bf16(a1, b, acc[1][t]);
        }
    }

#pragma unroll
    for (int t = 0; t < 4; ++t) {
        float bv = bias[n0 + t * 16 + l15];
#pragma unroll
        for (int rt = 0; rt < 2; ++rt) {
#pragma unroll
            for (int r = 0; r < 8; ++r) {
                int row = m0 + rt * 16 + r + half * 8;
                out[(size_t)row * DMODEL + n0 + t * 16 + l15] =
                    acc[rt][t][r] + bv;
            }
        }
    }
}

// ---------------------------------------------------------------------------
// Host-side launcher
// ---------------------------------------------------------------------------
extern "C" void kernel_launch(void* const* d_in, const int* in_sizes, int n_in,
                              void* d_out, int out_size, void* d_ws, size_t ws_size,
                              hipStream_t stream) {
    const float* x     = (const float*)d_in[0];   // [2,2048,1024]
    const float* w_qkv = (const float*)d_in[1];   // [3072,1024]
    const float* w_out = (const float*)d_in[2];   // [1024,1024]
    const float* b_out = (const float*)d_in[3];   // [1024]
    float* out = (float*)d_out;                   // [2,2048,1024] fp32

    char* ws = (char*)d_ws;
    size_t off = 0;
    bf16_t* xb    = (bf16_t*)(ws + off); off += (size_t)MTOT * DMODEL * 2;       // 8 MiB
    bf16_t* wqkvb = (bf16_t*)(ws + off); off += (size_t)3 * DMODEL * DMODEL * 2; // 6 MiB
    bf16_t* woutb = (bf16_t*)(ws + off); off += (size_t)DMODEL * DMODEL * 2;     // 2 MiB
    bf16_t* qb    = (bf16_t*)(ws + off); off += (size_t)MTOT * DMODEL * 2;       // 8 MiB
    bf16_t* kb    = (bf16_t*)(ws + off); off += (size_t)MTOT * DMODEL * 2;       // 8 MiB
    bf16_t* vtb   = (bf16_t*)(ws + off); off += (size_t)MTOT * DMODEL * 2;       // 8 MiB
    bf16_t* aob   = (bf16_t*)(ws + off); off += (size_t)MTOT * DMODEL * 2;       // 8 MiB

    // Stage 1: casts
    cast_f32_bf16<<<2048, 256, 0, stream>>>(x, xb, MTOT * DMODEL);
    cast_f32_bf16<<<2048, 256, 0, stream>>>(w_qkv, wqkvb, 3 * DMODEL * DMODEL);
    cast_f32_bf16<<<1024, 256, 0, stream>>>(w_out, woutb, DMODEL * DMODEL);

    // Stage 2: QKV projection (grid: 4096/32 x 3072/64)
    qkv_gemm<<<dim3(MTOT / 32, (3 * DMODEL) / 64), 32, 0, stream>>>(
        xb, wqkvb, qb, kb, vtb);

    // Stage 3: flash attention (grid: 2048/32 x heads x batch)
    attn_kernel<<<dim3(NSEQ / 32, NHEADS, BATCH), 32, 0, stream>>>(
        qb, kb, vtb, aob);

    // Stage 4: output projection + bias (grid: 4096/32 x 1024/64)
    out_gemm<<<dim3(MTOT / 32, DMODEL / 64), 32, 0, stream>>>(
        aob, woutb, b_out, out);
}
